// MemEffAttention_60816736911486
// MI455X (gfx1250) — matrix operand order repaired and checked by reading, not compiled, not run on hardware
//
#include <hip/hip_runtime.h>
#include <hip/hip_bf16.h>

typedef _Float16 v16h __attribute__((ext_vector_type(16)));
typedef _Float16 v8h  __attribute__((ext_vector_type(8)));
typedef float    v8f  __attribute__((ext_vector_type(8)));

union U16 { v16h v; v8h h[2]; };

__device__ __forceinline__ v8h shfl16_v8h(v8h x) {
    union { v8h h; int i[4]; } u; u.h = x;
#pragma unroll
    for (int j = 0; j < 4; ++j) u.i[j] = __shfl_xor(u.i[j], 16, 32);
    return u.h;
}

// ---------------------------------------------------------------- f32 -> f16
__global__ __launch_bounds__(256)
void f32_to_f16_kernel(const float* __restrict__ in, _Float16* __restrict__ out, int n) {
    int i = blockIdx.x * 256 + threadIdx.x;
    if (i < n) out[i] = (_Float16)in[i];
}

// ------------------------------------------------- C[M,N] = A[M,K] @ B[N,K]^T
// Register-blocked: each wave computes a 64x64 tile = 4x4 WMMA tiles.
// Per K-step: 8 fragment loads feed 16 v_wmma_f32_16x16x32_f16 (1 b128/wmma).
__global__ __launch_bounds__(128)
void wmma_gemm_nt(const _Float16* __restrict__ A, const _Float16* __restrict__ Bm,
                  float* __restrict__ C, const float* __restrict__ bias,
                  int M, int N, int K) {
    const int wave = threadIdx.x >> 5;
    const int lane = threadIdx.x & 31;
    const int job  = blockIdx.x * 4 + wave;
    const int nblk = N >> 6;
    const int bm   = job / nblk;
    const int bn   = job - bm * nblk;
    if (bm >= (M >> 6)) return;                 // wave-uniform guard

    const int lq  = lane & 15;
    const bool hi = (lane & 16) != 0;
    const int m0  = bm << 6;
    const int n0  = bn << 6;

    // A fragment row ptr: lane -> row; lo lanes K{0..7,16..23}, hi lanes K{8..15,24..31}
    const _Float16* aBase = A  + (size_t)(m0 + lq) * K + (hi ? 8 : 0);
    // B fragment: lane -> column of C (= row of Bm); lo lanes K{0..7,16..23}, hi lanes K{8..15,24..31}
    const _Float16* bBase = Bm + (size_t)(n0 + lq) * K + (hi ? 8 : 0);
    const size_t rstep = (size_t)16 * K;

    v8f acc[4][4] = {};
    for (int k0 = 0; k0 < K; k0 += 32) {
        U16 af[4], bf[4];
#pragma unroll
        for (int i = 0; i < 4; ++i) {
            const _Float16* p = aBase + (size_t)i * rstep + k0;
            af[i].h[0] = *(const v8h*)(p);
            af[i].h[1] = *(const v8h*)(p + 16);
            __builtin_prefetch(p + 128, 0, 1);
        }
#pragma unroll
        for (int j = 0; j < 4; ++j) {
            const _Float16* p = bBase + (size_t)j * rstep + k0;
            bf[j].h[0] = *(const v8h*)(p);
            bf[j].h[1] = *(const v8h*)(p + 16);
            __builtin_prefetch(p + 128, 0, 1);
        }
#pragma unroll
        for (int i = 0; i < 4; ++i)
#pragma unroll
            for (int j = 0; j < 4; ++j)
                acc[i][j] = __builtin_amdgcn_wmma_f32_16x16x32_f16(
                    false, af[i].v, false, bf[j].v, (short)0, acc[i][j], false, false);
    }

    const int mrow = m0 + (hi ? 8 : 0);
#pragma unroll
    for (int j = 0; j < 4; ++j) {
        const int ncol = n0 + j * 16 + lq;
        const float bn_v = bias ? bias[ncol] : 0.0f;
#pragma unroll
        for (int i = 0; i < 4; ++i) {
#pragma unroll
            for (int r = 0; r < 8; ++r)
                C[(size_t)(mrow + i * 16 + r) * N + ncol] = acc[i][j][r] + bn_v;
        }
    }
}

// ----------------------------------------------------- RoPE + layout scatter
// qkv: fp32 [B*L, 3*C]; writes q,k as f16 [B*H, L, D]; v transposed [B*H, D, L]
__global__ __launch_bounds__(256)
void rope_scatter(const float* __restrict__ qkv, const float* __restrict__ cosT,
                  const float* __restrict__ sinT,
                  _Float16* __restrict__ qh, _Float16* __restrict__ kh,
                  _Float16* __restrict__ vt) {
    const int idx = blockIdx.x * 256 + threadIdx.x;     // (b,l,h,p) p in [0,32)
    const int p = idx & 31;
    const int h = (idx >> 5) & 15;
    const int l = (idx >> 9) & 2047;
    const int b = idx >> 20;

    const size_t row = ((size_t)b * 2048 + l) * 3072;
    const int col = h * 64 + 2 * p;
    const float qe = qkv[row + col],        qo = qkv[row + col + 1];
    const float ke = qkv[row + 1024 + col], ko = qkv[row + 1024 + col + 1];
    const float ve = qkv[row + 2048 + col], vo = qkv[row + 2048 + col + 1];
    const float c = cosT[l * 32 + p], s = sinT[l * 32 + p];

    const size_t bh = (size_t)b * 16 + h;
    const size_t qkrow = (bh * 2048 + l) * 64 + 2 * p;
    qh[qkrow]     = (_Float16)(qe * c - qo * s);
    qh[qkrow + 1] = (_Float16)(qe * s + qo * c);
    kh[qkrow]     = (_Float16)(ke * c - ko * s);
    kh[qkrow + 1] = (_Float16)(ke * s + ko * c);
    vt[(bh * 64 + 2 * p)     * 2048 + l] = (_Float16)ve;
    vt[(bh * 64 + 2 * p + 1) * 2048 + l] = (_Float16)vo;
}

// ---------------------------------------------------------- flash attention
// One wave per (b,h, 32-query block): two 16-query tiles share every K and V
// fragment. S^T = K_tile @ Q^T, O^T = V^T @ P^T. 16 WMMA per 32-key step.
__global__ __launch_bounds__(256)
void attn_wmma(const _Float16* __restrict__ qh, const _Float16* __restrict__ kh,
               const _Float16* __restrict__ vt, _Float16* __restrict__ attn) {
    const int wave = threadIdx.x >> 5;
    const int lane = threadIdx.x & 31;
    const int wid  = blockIdx.x * 8 + wave;     // 2048 waves total
    const int l0   = (wid & 63) << 5;           // 32-query block start
    const int bh   = wid >> 6;                  // 0..31
    const int lq   = lane & 15;
    const bool hi  = (lane & 16) != 0;

    // Q^T B-fragments: [query-tile][K-step(d 0..31 / 32..63)]
    U16 qB[2][2];
#pragma unroll
    for (int qt = 0; qt < 2; ++qt) {
        const _Float16* qrow =
            qh + ((size_t)bh * 2048 + l0 + qt * 16 + lq) * 64 + (hi ? 8 : 0);
        qB[qt][0].h[0] = *(const v8h*)(qrow);
        qB[qt][0].h[1] = *(const v8h*)(qrow + 16);
        qB[qt][1].h[0] = *(const v8h*)(qrow + 32);
        qB[qt][1].h[1] = *(const v8h*)(qrow + 48);
    }

    v8f o[2][4] = {};
    float mrun[2] = {-1e30f, -1e30f};
    float lrun[2] = {0.0f, 0.0f};
    const float scale = 0.125f;   // D^-0.5, D=64

    const _Float16* kbase = kh + (size_t)bh * 2048 * 64;
    const _Float16* vbase = vt + (size_t)bh * 64 * 2048;

    for (int m0 = 0; m0 < 2048; m0 += 32) {
        // ---- K A-fragments (shared by both query tiles)
        const _Float16* kr0 = kbase + (size_t)(m0 + lq) * 64 + (hi ? 8 : 0);
        const _Float16* kr1 = kr0 + 16 * 64;
        U16 ka00, ka01, ka10, ka11;
        ka00.h[0] = *(const v8h*)(kr0);      ka00.h[1] = *(const v8h*)(kr0 + 16);
        ka01.h[0] = *(const v8h*)(kr0 + 32); ka01.h[1] = *(const v8h*)(kr0 + 48);
        ka10.h[0] = *(const v8h*)(kr1);      ka10.h[1] = *(const v8h*)(kr1 + 16);
        ka11.h[0] = *(const v8h*)(kr1 + 32); ka11.h[1] = *(const v8h*)(kr1 + 48);

        // ---- V^T A-fragments (shared by both query tiles)
        const _Float16* vr = vbase + m0 + (hi ? 16 : 0);
        U16 va[4];
#pragma unroll
        for (int t = 0; t < 4; ++t) {
            const _Float16* vrow = vr + (size_t)(t * 16 + lq) * 2048;
            va[t].h[0] = *(const v8h*)(vrow);
            va[t].h[1] = *(const v8h*)(vrow + 8);
        }

        U16 pB[2];
#pragma unroll
        for (int qt = 0; qt < 2; ++qt) {
            // ---- S^T (32 keys x 16 queries) via 4 WMMA
            v8f st0 = {}, st1 = {};
            st0 = __builtin_amdgcn_wmma_f32_16x16x32_f16(false, ka00.v, false, qB[qt][0].v, (short)0, st0, false, false);
            st0 = __builtin_amdgcn_wmma_f32_16x16x32_f16(false, ka01.v, false, qB[qt][1].v, (short)0, st0, false, false);
            st1 = __builtin_amdgcn_wmma_f32_16x16x32_f16(false, ka10.v, false, qB[qt][0].v, (short)0, st1, false, false);
            st1 = __builtin_amdgcn_wmma_f32_16x16x32_f16(false, ka11.v, false, qB[qt][1].v, (short)0, st1, false, false);

            // ---- online softmax (lane = query; partner lane L^16 has other 16 keys)
            float vmax = -1e30f;
#pragma unroll
            for (int r = 0; r < 8; ++r) {
                st0[r] *= scale; st1[r] *= scale;
                vmax = fmaxf(vmax, fmaxf(st0[r], st1[r]));
            }
            vmax = fmaxf(vmax, __shfl_xor(vmax, 16, 32));
            const float mnew  = fmaxf(mrun[qt], vmax);
            const float alpha = __expf(mrun[qt] - mnew);

            float ssum = 0.0f;
            v8h x0, x1;
#pragma unroll
            for (int r = 0; r < 8; ++r) {
                const float p0 = __expf(st0[r] - mnew);
                const float p1 = __expf(st1[r] - mnew);
                ssum += p0 + p1;
                x0[r] = (_Float16)p0;
                x1[r] = (_Float16)p1;
            }
            ssum += __shfl_xor(ssum, 16, 32);
            lrun[qt] = lrun[qt] * alpha + ssum;
            mrun[qt] = mnew;
#pragma unroll
            for (int t = 0; t < 4; ++t)
#pragma unroll
                for (int r = 0; r < 8; ++r) o[qt][t][r] *= alpha;

            // ---- rearrange S^T D-layout -> P^T B-fragment (keys along K dim)
            const v8h y0 = shfl16_v8h(x0);
            const v8h y1 = shfl16_v8h(x1);
            pB[qt].h[0] = hi ? y1 : x0;   // keys {0..7} | {16..23}
            pB[qt].h[1] = hi ? x1 : y0;   // keys {8..15} | {24..31}
        }

        // ---- O^T += V^T @ P^T (V fragments reused across both query tiles)
#pragma unroll
        for (int t = 0; t < 4; ++t) {
#pragma unroll
            for (int qt = 0; qt < 2; ++qt)
                o[qt][t] = __builtin_amdgcn_wmma_f32_16x16x32_f16(
                    false, va[t].v, false, pB[qt].v, (short)0, o[qt][t], false, false);
        }
    }

    // ---- finalize: divide by row sum, store f16 attn[b, l, h*64+d]
    const int b = bh >> 4, h = bh & 15;
#pragma unroll
    for (int qt = 0; qt < 2; ++qt) {
        const float inv = 1.0f / lrun[qt];
        _Float16* orow = attn + (size_t)(b * 2048 + l0 + qt * 16 + lq) * 1024
                              + h * 64 + (hi ? 8 : 0);
#pragma unroll
        for (int t = 0; t < 4; ++t) {
            v8h oh;
#pragma unroll
            for (int r = 0; r < 8; ++r) oh[r] = (_Float16)(o[qt][t][r] * inv);
            *(v8h*)(orow + t * 16) = oh;
        }
    }
}

// ---------------------------------------------------------------------------
extern "C" void kernel_launch(void* const* d_in, const int* in_sizes, int n_in,
                              void* d_out, int out_size, void* d_ws, size_t ws_size,
                              hipStream_t stream) {
    const float* x      = (const float*)d_in[0];   // [2,2048,1024]
    const float* qkv_w  = (const float*)d_in[1];   // [3072,1024]
    const float* proj_w = (const float*)d_in[2];   // [1024,1024]
    const float* proj_b = (const float*)d_in[3];   // [1024]
    const float* cosT   = (const float*)d_in[4];   // [2048,32]
    const float* sinT   = (const float*)d_in[5];   // [2048,32]
    float* out = (float*)d_out;                    // [2,2048,1024] fp32

    char* w = (char*)d_ws;
    _Float16* x_h     = (_Float16*)w; w += (size_t)8  << 20;  // 4096x1024 f16
    _Float16* qkvw_h  = (_Float16*)w; w += (size_t)6  << 20;  // 3072x1024 f16
    _Float16* projw_h = (_Float16*)w; w += (size_t)2  << 20;  // 1024x1024 f16
    float*    qkv_f   = (float*)w;    w += (size_t)48 << 20;  // 4096x3072 f32
    _Float16* qh      = (_Float16*)w; w += (size_t)8  << 20;  // [32,2048,64]
    _Float16* kh      = (_Float16*)w; w += (size_t)8  << 20;  // [32,2048,64]
    _Float16* vtr     = (_Float16*)w; w += (size_t)8  << 20;  // [32,64,2048]
    _Float16* attn_h  = (_Float16*)w; w += (size_t)8  << 20;  // 4096x1024 f16

    // 1) fp32 -> f16 conversions
    f32_to_f16_kernel<<<4194304 / 256, 256, 0, stream>>>(x,      x_h,     4194304);
    f32_to_f16_kernel<<<3145728 / 256, 256, 0, stream>>>(qkv_w,  qkvw_h,  3145728);
    f32_to_f16_kernel<<<1048576 / 256, 256, 0, stream>>>(proj_w, projw_h, 1048576);

    // 2) QKV GEMM: [4096,1024] @ [3072,1024]^T -> fp32 [4096,3072]
    //    64x64 tile per wave: (4096/64)*(3072/64) = 3072 jobs, 4 waves/block
    wmma_gemm_nt<<<768, 128, 0, stream>>>(x_h, qkvw_h, qkv_f, nullptr, 4096, 3072, 1024);

    // 3) RoPE + head-major scatter (q,k) and V transpose
    rope_scatter<<<8192, 256, 0, stream>>>(qkv_f, cosT, sinT, qh, kh, vtr);

    // 4) flash attention, one wave per (b,h, 32-query block): 2048 waves
    attn_wmma<<<256, 256, 0, stream>>>(qh, kh, vtr, attn_h);

    // 5) output projection + bias -> fp32 d_out: (4096/64)*(1024/64) = 1024 jobs
    wmma_gemm_nt<<<256, 128, 0, stream>>>(attn_h, projw_h, out, proj_b, 4096, 1024, 1024);
}
